// ATTLayer_24197845745848
// MI455X (gfx1250) — matrix operand order repaired and checked by reading, not compiled, not run on hardware
//
#include <hip/hip_runtime.h>
#include <hip/hip_fp16.h>

// Shapes from the reference
#define NB 4
#define NC 64
#define NF 256
#define NT 256
#define NCH 32
#define EPSV 1e-5f

typedef _Float16 half16 __attribute__((ext_vector_type(16)));
typedef _Float16 half8  __attribute__((ext_vector_type(8)));
typedef _Float16 half2v __attribute__((ext_vector_type(2)));
typedef float    float8 __attribute__((ext_vector_type(8)));

// ---------------------------------------------------------------------------
// WMMA 1x1 conv + BN + LeakyReLU.  A 1x1 conv is a GEMM:
//     Out[pix][o] = lrelu(BN( sum_i X[pix][i] * W[o][i] ))
// Each block owns 256 consecutive pixels of one batch; activations are staged
// to LDS as [pix][cin] f16 and weights as [cout][cin] f16 so both WMMA
// operands are contiguous 16B/32B ds_load_b128 fragments (K = Cin, 32/WMMA).
//   MODE 0: fqk  (Cin=64,Cout=64, ch-major f32 in, Q/K de-interleave, u=t,w=f)
//   MODE 1: tqk  (same, u=f,w=t)
//   MODE 2: fv   (Cin=64,Cout=32, ch-major f32 in, V transposed [b][t][c][f])
//   MODE 3: proj (Cin=32,Cout=64, pixel-major f16 in, f32 out + residual)
// ---------------------------------------------------------------------------
template <int MODE>
__global__ __launch_bounds__(256) void conv_wmma_kernel(
    const float* __restrict__ Xf32, const _Float16* __restrict__ Xf16,
    const float* __restrict__ W, const float* __restrict__ bb,
    const float* __restrict__ gg, const float* __restrict__ bee,
    const float* __restrict__ mm, const float* __restrict__ vv,
    const float* __restrict__ aa,
    _Float16* __restrict__ O0, _Float16* __restrict__ O1,
    const float* __restrict__ Xres, float* __restrict__ OutF) {
  constexpr int CIN  = (MODE == 3) ? 32 : 64;
  constexpr int COUT = (MODE == 2) ? 32 : 64;
  constexpr int NKC  = CIN / 32;    // K chunks per WMMA row
  constexpr int NCT  = COUT / 16;   // output column tiles

  __shared__ _Float16 sX[256 * CIN];     // activations [pix][i]
  __shared__ _Float16 sW[COUT * CIN];    // weights     [o][i]

  const int tid = threadIdx.x;
  const int b = blockIdx.x >> 8;                  // 256 blocks per batch
  const int pbase = (blockIdx.x & 255) * 256;     // pixel base within batch

  // ---- stage activations ----
  if (MODE == 3) {
    // pixel-major f16 input: straight 128-bit copy (256*32 halves = 16KB)
    const float4* s = (const float4*)(Xf16 + (size_t)blockIdx.x * 256 * CIN);
    float4* d = (float4*)sX;
#pragma unroll
    for (int i = 0; i < 4; ++i) d[tid + i * 256] = s[tid + i * 256];
  } else {
    // channel-major f32 input: each thread owns one pixel, packs f16 pairs
    const int pix = pbase + tid;
    const float* Xb = Xf32 + (size_t)b * CIN * (NF * NT) + pix;
#pragma unroll
    for (int i = 0; i < CIN; i += 2) {
      _Float16 h0 = (_Float16)Xb[(size_t)i * (NF * NT)];
      _Float16 h1 = (_Float16)Xb[(size_t)(i + 1) * (NF * NT)];
      *(half2v*)&sX[tid * CIN + i] = (half2v){h0, h1};
    }
  }
  // ---- stage weights ----
#pragma unroll
  for (int j = tid; j < COUT * CIN; j += 256) sW[j] = (_Float16)W[j];
  __syncthreads();

  const int lane = tid & 31;
  const int wave = tid >> 5;
  const int lo = lane & 15;
  const int hi = lane >> 4;
  const float alpha = aa[0];

  for (int rt = wave; rt < 16; rt += 8) {   // row (pixel) tile, uniform/wave
    // A fragments (reused across all column tiles)
    half16 af[NKC];
#pragma unroll
    for (int kc = 0; kc < NKC; ++kc) {
      const half8* xrow = (const half8*)&sX[(rt * 16 + lo) * CIN + 32 * kc];
      half8 a0 = xrow[2 * hi];
      half8 a1 = xrow[2 * hi + 1];
      af[kc] = __builtin_shufflevector(a0, a1, 0, 1, 2, 3, 4, 5, 6, 7,
                                       8, 9, 10, 11, 12, 13, 14, 15);
    }
#pragma unroll
    for (int ct = 0; ct < NCT; ++ct) {
      float8 acc;
#pragma unroll
      for (int r = 0; r < 8; ++r) acc[r] = 0.f;
#pragma unroll
      for (int kc = 0; kc < NKC; ++kc) {
        half16 bf = *(const half16*)&sW[(ct * 16 + lo) * CIN + 32 * kc + 16 * hi];
        acc = __builtin_amdgcn_wmma_f32_16x16x32_f16(
            false, af[kc], false, bf, (short)0, acc, false, false);
      }
      // BN + LeakyReLU epilogue (per output channel o = column)
      const int o = ct * 16 + lo;
      const float sc = gg[o] * rsqrtf(vv[o] + EPSV);
      const float ofs = bb[o] - mm[o];
      const float beo = bee[o];
#pragma unroll
      for (int r = 0; r < 8; ++r) {
        float y = (acc[r] + ofs) * sc + beo;
        y = y > 0.f ? y : alpha * y;
        const int pb = pbase + rt * 16 + r + 8 * hi;  // pixel within batch
        const int f = pb >> 8, t = pb & 255;
        if (MODE == 0) {
          const int c = o >> 1;
          size_t adr = (((size_t)b * 256 + t) * 256 + f) * NCH + c;
          if ((o & 1) == 0) O0[adr] = (_Float16)y; else O1[adr] = (_Float16)y;
        } else if (MODE == 1) {
          const int c = o >> 1;
          size_t adr = (((size_t)b * 256 + f) * 256 + t) * NCH + c;
          if ((o & 1) == 0) O0[adr] = (_Float16)y; else O1[adr] = (_Float16)y;
        } else if (MODE == 2) {
          O0[(((size_t)b * 256 + t) * NCH + o) * NF + f] = (_Float16)y;
        } else {
          size_t idx = ((size_t)(b * NC + o) * NF + f) * NT + t;
          OutF[idx] = y + Xres[idx];
        }
      }
    }
  }
}

// ---------------------------------------------------------------------------
// WMMA attention over one 256x256 strip: S = Q(256x32) * K^T(32x256),
// (optional causal mask,) softmax rows, O = P(256x256) * V(256x32).
// One block per strip, 8 waves, each wave owns 2 row-tiles of 16 rows.
// ---------------------------------------------------------------------------
template <bool CAUSAL>
__global__ __launch_bounds__(256) void attn_kernel(
    const _Float16* __restrict__ Qb, const _Float16* __restrict__ Kb,
    const _Float16* __restrict__ Vb, _Float16* __restrict__ Ob,
    int obaseMul, int rowMul, int colMul) {
  __shared__ _Float16 sQ[NF * NCH];     // 16 KB  [row][c]
  __shared__ _Float16 sK[NF * NCH];     // 16 KB  [col][c]   (K un-transposed)
  __shared__ _Float16 sVT[NCH * NF];    // 16 KB  [c][y]     (V transposed)
  __shared__ _Float16 sP[8][16 * 32];   //  8 KB  per-wave P-chunk relayout

  const int strip = blockIdx.x;                     // b*256 + (t or f)
  const _Float16* Qg = Qb + (size_t)strip * (NF * NCH);
  const _Float16* Kg = Kb + (size_t)strip * (NF * NCH);
  const _Float16* Vg = Vb + (size_t)strip * (NCH * NF);
  _Float16* Og = Ob + (size_t)(strip >> 8) * (NF * NT * NCH) +
                 (size_t)(strip & 255) * obaseMul;

  __builtin_prefetch(Qg, 0, 1);   // global_prefetch_b8
  __builtin_prefetch(Kg, 0, 1);
  __builtin_prefetch(Vg, 0, 1);

  {
    const float4* q4 = (const float4*)Qg;
    const float4* k4 = (const float4*)Kg;
    const float4* v4 = (const float4*)Vg;
    float4* dq = (float4*)sQ;
    float4* dk = (float4*)sK;
    float4* dv = (float4*)sVT;
#pragma unroll
    for (int i = 0; i < 4; ++i) {
      int idx = threadIdx.x + i * 256;
      dq[idx] = q4[idx];
      dk[idx] = k4[idx];
      dv[idx] = v4[idx];
    }
  }
  __syncthreads();

  const int lane = threadIdx.x & 31;
  const int wave = threadIdx.x >> 5;
  const int lo = lane & 15;
  const int hi = lane >> 4;
  const float scale = 0.17677669529663687f;  // 1/sqrt(32)

  for (int rt = wave; rt < 16; rt += 8) {    // row tile (uniform per wave)
    half16 aq;
    {
      const half8* qrow = (const half8*)&sQ[(rt * 16 + lo) * NCH];
      half8 a0 = qrow[2 * hi];
      half8 a1 = qrow[2 * hi + 1];
      aq = __builtin_shufflevector(a0, a1, 0, 1, 2, 3, 4, 5, 6, 7,
                                   8, 9, 10, 11, 12, 13, 14, 15);
    }

    float8 acc[16];
#pragma unroll
    for (int kt = 0; kt < 16; ++kt)
#pragma unroll
      for (int r = 0; r < 8; ++r) acc[kt][r] = 0.f;

#pragma unroll
    for (int kt = 0; kt < 16; ++kt) {
      if (!CAUSAL || kt <= rt) {             // uniform guard: EXEC stays full
        half16 bk = *(const half16*)&sK[(kt * 16 + lo) * NCH + 16 * hi];
        acc[kt] = __builtin_amdgcn_wmma_f32_16x16x32_f16(
            false, aq, false, bk, (short)0, acc[kt], false, false);
        if (CAUSAL && kt == rt) {
#pragma unroll
          for (int r = 0; r < 8; ++r) {
            int row = rt * 16 + r + 8 * hi;
            int col = kt * 16 + lo;
            if (col > row) acc[kt][r] = -1e30f;
          }
        }
      }
    }

    // softmax over the 256-wide row (scaled)
    float mrow[8], srow[8];
#pragma unroll
    for (int r = 0; r < 8; ++r) mrow[r] = -1e30f;
#pragma unroll
    for (int kt = 0; kt < 16; ++kt)
      if (!CAUSAL || kt <= rt)
#pragma unroll
        for (int r = 0; r < 8; ++r) mrow[r] = fmaxf(mrow[r], acc[kt][r]);
#pragma unroll
    for (int r = 0; r < 8; ++r) {
      float m = mrow[r];
      m = fmaxf(m, __shfl_xor(m, 1, 16));
      m = fmaxf(m, __shfl_xor(m, 2, 16));
      m = fmaxf(m, __shfl_xor(m, 4, 16));
      m = fmaxf(m, __shfl_xor(m, 8, 16));
      mrow[r] = m;
      srow[r] = 0.f;
    }
#pragma unroll
    for (int kt = 0; kt < 16; ++kt)
      if (!CAUSAL || kt <= rt)
#pragma unroll
        for (int r = 0; r < 8; ++r) {
          float e = __expf((acc[kt][r] - mrow[r]) * scale);
          acc[kt][r] = e;
          srow[r] += e;
        }
#pragma unroll
    for (int r = 0; r < 8; ++r) {
      float s = srow[r];
      s += __shfl_xor(s, 1, 16);
      s += __shfl_xor(s, 2, 16);
      s += __shfl_xor(s, 4, 16);
      s += __shfl_xor(s, 8, 16);
      srow[r] = 1.0f / s;
    }
#pragma unroll
    for (int kt = 0; kt < 16; ++kt)
      if (!CAUSAL || kt <= rt)
#pragma unroll
        for (int r = 0; r < 8; ++r) acc[kt][r] *= srow[r];

    // O = P * V : per 32-wide K chunk, relayout P (C->A) via LDS
    float8 oacc0, oacc1;
#pragma unroll
    for (int r = 0; r < 8; ++r) { oacc0[r] = 0.f; oacc1[r] = 0.f; }

#pragma unroll
    for (int kc = 0; kc < 8; ++kc) {
      if (CAUSAL && 2 * kc > rt) continue;
      __builtin_amdgcn_wave_barrier();
#pragma unroll
      for (int s2 = 0; s2 < 2; ++s2) {
        int kt = 2 * kc + s2;
#pragma unroll
        for (int r = 0; r < 8; ++r) {
          float val = (!CAUSAL || kt <= rt) ? acc[kt][r] : 0.f;
          sP[wave][(r + 8 * hi) * 32 + s2 * 16 + lo] = (_Float16)val;
        }
      }
      __builtin_amdgcn_wave_barrier();
      asm volatile("s_wait_dscnt 0" ::: "memory");   // in-wave LDS RAW

      half16 ap;
      {
        const half8* prow = (const half8*)&sP[wave][lo * 32];
        half8 p0 = prow[2 * hi];
        half8 p1 = prow[2 * hi + 1];
        ap = __builtin_shufflevector(p0, p1, 0, 1, 2, 3, 4, 5, 6, 7,
                                     8, 9, 10, 11, 12, 13, 14, 15);
      }
      half16 bv0 = *(const half16*)&sVT[lo * NF + 32 * kc + 16 * hi];
      half16 bv1 = *(const half16*)&sVT[(16 + lo) * NF + 32 * kc + 16 * hi];

      oacc0 = __builtin_amdgcn_wmma_f32_16x16x32_f16(
          false, ap, false, bv0, (short)0, oacc0, false, false);
      oacc1 = __builtin_amdgcn_wmma_f32_16x16x32_f16(
          false, ap, false, bv1, (short)0, oacc1, false, false);
    }

#pragma unroll
    for (int r = 0; r < 8; ++r) {
      int row = rt * 16 + r + 8 * hi;
      Og[(size_t)row * rowMul + (size_t)lo * colMul]        = (_Float16)oacc0[r];
      Og[(size_t)row * rowMul + (size_t)(16 + lo) * colMul] = (_Float16)oacc1[r];
    }
  }
}

// ---------------------------------------------------------------------------
extern "C" void kernel_launch(void* const* d_in, const int* in_sizes, int n_in,
                              void* d_out, int out_size, void* d_ws,
                              size_t ws_size, hipStream_t stream) {
  (void)in_sizes; (void)n_in; (void)out_size; (void)ws_size;

  const float* inp = (const float*)d_in[0];
  const float* x   = (const float*)d_in[1];
  const float* fqk[7]  = {(const float*)d_in[2],  (const float*)d_in[3],
                          (const float*)d_in[4],  (const float*)d_in[5],
                          (const float*)d_in[6],  (const float*)d_in[7],
                          (const float*)d_in[8]};
  const float* fv[7]   = {(const float*)d_in[9],  (const float*)d_in[10],
                          (const float*)d_in[11], (const float*)d_in[12],
                          (const float*)d_in[13], (const float*)d_in[14],
                          (const float*)d_in[15]};
  const float* tqk[7]  = {(const float*)d_in[16], (const float*)d_in[17],
                          (const float*)d_in[18], (const float*)d_in[19],
                          (const float*)d_in[20], (const float*)d_in[21],
                          (const float*)d_in[22]};
  const float* proj[7] = {(const float*)d_in[23], (const float*)d_in[24],
                          (const float*)d_in[25], (const float*)d_in[26],
                          (const float*)d_in[27], (const float*)d_in[28],
                          (const float*)d_in[29]};
  float* out = (float*)d_out;

  // Workspace layout (needs 112 MB; fp16 buffers 16 MB each)
  const size_t HB = (size_t)NB * NT * NF * NCH * sizeof(_Float16);  // 16 MB
  char* ws = (char*)d_ws;
  _Float16* Qf   = (_Float16*)(ws + 0 * HB);   // [b][t][f][c]
  _Float16* Kf   = (_Float16*)(ws + 1 * HB);   // [b][t][f][c]
  _Float16* Vft  = (_Float16*)(ws + 2 * HB);   // [b][t][c][f]   (transposed)
  _Float16* Qt   = (_Float16*)(ws + 3 * HB);   // [b][f][t][c]
  _Float16* Kt   = (_Float16*)(ws + 4 * HB);   // [b][f][t][c]
  _Float16* Fout = (_Float16*)(ws + 5 * HB);   // [b][f][c][t]   (transposed)
  _Float16* Tout = (_Float16*)(ws + 6 * HB);   // [b][f][t][c]

  dim3 grid(NB * 256), blk(256);               // 256 pixels per block
  dim3 agrid(NB * NT);                         // 1024 strips

  // q/k/v projections as WMMA GEMMs
  conv_wmma_kernel<0><<<grid, blk, 0, stream>>>(
      inp, nullptr, fqk[0], fqk[1], fqk[2], fqk[3], fqk[4], fqk[5], fqk[6],
      Qf, Kf, nullptr, nullptr);
  conv_wmma_kernel<2><<<grid, blk, 0, stream>>>(
      x, nullptr, fv[0], fv[1], fv[2], fv[3], fv[4], fv[5], fv[6],
      Vft, nullptr, nullptr, nullptr);
  conv_wmma_kernel<1><<<grid, blk, 0, stream>>>(
      inp, nullptr, tqk[0], tqk[1], tqk[2], tqk[3], tqk[4], tqk[5], tqk[6],
      Qt, Kt, nullptr, nullptr);

  // frequency attention: strip=(b,t); out Fout[b][f][c][t] fp16 (transposed,
  // so it is directly the [c][y] V layout for the time attention)
  attn_kernel<false><<<agrid, blk, 0, stream>>>(
      Qf, Kf, Vft, Fout,
      /*obaseMul=*/1, /*rowMul=*/NCH * NT, /*colMul=*/NT);

  // causal time attention: strip=(b,f); V = Fout; out Tout[b][f][t][c] fp16
  attn_kernel<true><<<agrid, blk, 0, stream>>>(
      Qt, Kt, Fout, Tout,
      /*obaseMul=*/NT * NCH, /*rowMul=*/NCH, /*colMul=*/1);

  // projection + residual as WMMA GEMM (f32 out)
  conv_wmma_kernel<3><<<grid, blk, 0, stream>>>(
      nullptr, Tout, proj[0], proj[1], proj[2], proj[3], proj[4], proj[5],
      proj[6], nullptr, nullptr, x, out);
}
